// GraphAutoEncoder_87969520156915
// MI455X (gfx1250) — matrix operand order repaired and checked by reading, not compiled, not run on hardware
//
#include <hip/hip_runtime.h>
#include <hip/hip_bf16.h>

// ---------------------------------------------------------------------------
// GraphAutoEncoder for MI455X (gfx1250, wave32, WMMA + TDM).
// Dense GEMMs: v_wmma_f32_16x16x32_f16 (f16 in, f32 accumulate).
// Weights are pre-packed once into the WMMA B-fragment layout (f16), then
// per-K-step tiles are DMA'd into LDS by the Tensor Data Mover
// (tensor_load_to_lds / s_wait_tensorcnt) with double buffering.
// Attention gather/scatter uses f32 global atomics (memory-bound part).
// ---------------------------------------------------------------------------

typedef __attribute__((ext_vector_type(16))) _Float16     v16h;
typedef __attribute__((ext_vector_type(8)))  float        v8f;
typedef __attribute__((ext_vector_type(4)))  unsigned int u32x4;
typedef __attribute__((ext_vector_type(8)))  int          i32x8;
typedef __attribute__((ext_vector_type(4)))  int          i32x4;

#define GEMM_RELU  1
#define GEMM_ACCUM 2

#if defined(__has_builtin)
#if __has_builtin(__builtin_amdgcn_tensor_load_to_lds) && \
    __has_builtin(__builtin_amdgcn_s_wait_tensorcnt)
#define HAVE_TDM 1
#endif
#endif

#ifdef HAVE_TDM
// TDM copy of one contiguous 4096-byte packed-weight chunk into LDS.
// D# per CDNA5 ISA:
//   group0 = {count=1, lds_addr, 57-bit global_addr, type=2}
//   group1 = {data_size=4B, huge tensor dims (no OOB clip), tile 1024x1}
// Does NOT wait; caller uses __builtin_amdgcn_s_wait_tensorcnt(0).
__device__ __forceinline__ void tdm_load_chunk(_Float16* ldsDst, const _Float16* gsrc)
{
    const unsigned lds_off = (unsigned)(size_t)(void*)ldsDst;  // addr[31:0] = LDS byte addr
    const unsigned long long ga = (unsigned long long)(size_t)gsrc;

    u32x4 g0;
    g0[0] = 1u;                                      // count=1 (valid), user mode
    g0[1] = lds_off;                                 // lds_addr[31:0]
    g0[2] = (unsigned)(ga & 0xFFFFFFFFull);          // global_addr[31:0]
    g0[3] = (unsigned)((ga >> 32) & 0x01FFFFFFull)   // global_addr[56:32]
          | (2u << 30);                              // type = 2 ("image")

    const unsigned dim0 = 1u << 30;                  // tensor_dim0 (no OOB)
    const unsigned dim1 = 1u << 30;                  // tensor_dim1 (no OOB)
    i32x8 g1;
    g1[0] = (int)(2u << 16);                         // wg_mask=0, data_size=2 (4B)
    g1[1] = (int)((dim0 & 0xFFFFu) << 16);           // tensor_dim0[15:0]
    g1[2] = (int)(((dim0 >> 16) & 0xFFFFu)           // tensor_dim0[31:16]
          |       ((dim1 & 0xFFFFu) << 16));         // tensor_dim1[15:0]
    g1[3] = (int)(((dim1 >> 16) & 0xFFFFu)           // tensor_dim1[31:16]
          |       (1024u << 16));                    // tile_dim0 = 1024 dwords
    g1[4] = 1;                                       // tile_dim1 = 1 (1-D copy)
    g1[5] = 1024;                                    // tensor_dim0_stride
    g1[6] = 0;
    g1[7] = 0;

    i32x4 z4 = {0, 0, 0, 0};
#if __clang_major__ >= 23
    i32x8 z8 = {0, 0, 0, 0, 0, 0, 0, 0};
    __builtin_amdgcn_tensor_load_to_lds(g0, g1, z4, z4, z8, 0);
#else
    __builtin_amdgcn_tensor_load_to_lds(g0, g1, z4, z4, 0);
#endif
}
#endif  // HAVE_TDM

// ------------------------- weight fragment packing -------------------------
// Pack W[K][Fo] (f32, row-major) into f16 WMMA B-fragment order:
//   chunk cidx = (kb/32)*(Fo/16) + nf   (512 halves = 1 KB each)
//   within chunk: lane l (0..31) holds 16 contiguous halves:
//     b[i] = W[kb + 16*(l/16) + i][nf*16 + (l%16)]
// so a lane's whole fragment is one aligned 32-byte load.
__global__ void pack_weight_kernel(const float* __restrict__ W,
                                   _Float16* __restrict__ Wp, int K, int Fo)
{
    int tid = blockIdx.x * blockDim.x + threadIdx.x;
    int total = K * Fo;
    if (tid >= total) return;
    int cidx = tid >> 9;            // / 512
    int r    = tid & 511;
    int l    = r >> 4;
    int i    = r & 15;
    int nfo  = Fo >> 4;
    int kbi  = cidx / nfo;
    int nf   = cidx - kbi * nfo;
    int k    = kbi * 32 + ((l >> 4) << 4) + i;
    int col  = nf * 16 + (l & 15);
    Wp[tid] = (_Float16)W[(size_t)k * Fo + col];
}

// ----------------------------- WMMA GEMM -----------------------------------
// C[N x Fo] = op( A[N x K] @ W[K x Fo] + bias (+ C if ACCUM) )
// K multiple of 32, Fo multiple of 64. A row-major f32; W pre-packed f16.
// Block = 128 threads = 4 waves stacked in M; block tile = 64 rows x 64 cols.
// Each wave: 16 rows x 64 cols = 4 accumulators; A fragment loaded once per
// K-step and reused 4x. Packed B chunk (4 KB) is TDM-staged into LDS with
// double buffering (DMA of block k+1 overlapped with WMMAs of block k).
__global__ __launch_bounds__(128) void gemm_wmma_kernel(
    const float* __restrict__ A, const _Float16* __restrict__ Wp,
    const float* __restrict__ bias, float* __restrict__ C,
    int Nrows, int K, int Fo, int flags)
{
    __shared__ __align__(32) _Float16 wtile[2][2048];   // 2 x 4 KB

    const int lane = threadIdx.x & 31;
    const int wave = threadIdx.x >> 5;
    const int half = lane >> 4;
    const int l16  = lane & 15;
    const int row0 = (blockIdx.x * 4 + wave) * 16;
    const int col0 = blockIdx.y * 64;
    const int nfo  = Fo >> 4;

    const int  m      = row0 + l16;
    const bool mvalid = (m < Nrows);
    const float* Arow = A + (size_t)(mvalid ? m : 0) * K;

    v8f cacc[4] = {{}, {}, {}, {}};
    const int nkb = K >> 5;

    auto chunk_src = [&](int kbi) {
        return Wp + ((size_t)kbi * nfo + (size_t)blockIdx.y * 4) * 512;
    };

    auto do_kstep = [&](const _Float16* tile, int kbi) {
        // ---- A fragment: two aligned 8-float runs per lane, cvt to f16 ----
        v16h a;
        {
            const int k0 = (kbi << 5) + half * 16;
            const float4* p0 = reinterpret_cast<const float4*>(Arow + k0);
            const float4* p1 = reinterpret_cast<const float4*>(Arow + k0 + 8);
            float4 f0 = p0[0], f1 = p0[1], f2 = p1[0], f3 = p1[1];
            if (!mvalid) {
                float4 z4 = {0.f, 0.f, 0.f, 0.f};
                f0 = z4; f1 = z4; f2 = z4; f3 = z4;
            }
            a[0]  = (_Float16)f0.x; a[1]  = (_Float16)f0.y;
            a[2]  = (_Float16)f0.z; a[3]  = (_Float16)f0.w;
            a[4]  = (_Float16)f1.x; a[5]  = (_Float16)f1.y;
            a[6]  = (_Float16)f1.z; a[7]  = (_Float16)f1.w;
            a[8]  = (_Float16)f2.x; a[9]  = (_Float16)f2.y;
            a[10] = (_Float16)f2.z; a[11] = (_Float16)f2.w;
            a[12] = (_Float16)f3.x; a[13] = (_Float16)f3.y;
            a[14] = (_Float16)f3.z; a[15] = (_Float16)f3.w;
        }
        // ---- 4 B fragments: one aligned 32B LDS read each ----
#pragma unroll
        for (int nf = 0; nf < 4; ++nf) {
            v16h b = *reinterpret_cast<const v16h*>(tile + nf * 512 + lane * 16);
            cacc[nf] = __builtin_amdgcn_wmma_f32_16x16x32_f16(
                           false, a, false, b, (short)0, cacc[nf], false, false);
        }
    };

#ifdef HAVE_TDM
    // --- preload K-block 0 ---
    if (wave == 0) {
        tdm_load_chunk(wtile[0], chunk_src(0));
        __builtin_amdgcn_s_wait_tensorcnt(0);
    }
    __syncthreads();
    for (int kbi = 0; kbi < nkb; ++kbi) {
        const int cur = kbi & 1;
        if (wave == 0 && kbi + 1 < nkb)
            tdm_load_chunk(wtile[1 - cur], chunk_src(kbi + 1));   // overlap DMA
        do_kstep(wtile[cur], kbi);
        if (wave == 0)
            __builtin_amdgcn_s_wait_tensorcnt(0);
        __syncthreads();
    }
#else
    for (int kbi = 0; kbi < nkb; ++kbi) {
        __syncthreads();
        {   // cooperative copy of the contiguous 4 KB chunk
            const unsigned int* src = (const unsigned int*)chunk_src(kbi);
            unsigned int* dst = (unsigned int*)wtile[0];
            for (int i = threadIdx.x; i < 1024; i += 128) dst[i] = src[i];
        }
        __syncthreads();
        do_kstep(wtile[0], kbi);
    }
#endif

    // ---- epilogue ----
#pragma unroll
    for (int nf = 0; nf < 4; ++nf) {
#pragma unroll
        for (int v = 0; v < 8; ++v) {
            const int row = row0 + v + 8 * half;
            if (row < Nrows) {
                const int col = col0 + nf * 16 + l16;
                const size_t off = (size_t)row * Fo + col;
                float val = cacc[nf][v];
                if (bias)               val += bias[col];
                if (flags & GEMM_ACCUM) val += C[off];
                if (flags & GEMM_RELU)  val = val > 0.f ? val : 0.f;
                C[off] = val;
            }
        }
    }
}

// ------------------------- small helper kernels ----------------------------

__device__ __forceinline__ unsigned int fkey(float f) {
    unsigned int u = __float_as_uint(f);
    return (u & 0x80000000u) ? ~u : (u | 0x80000000u);
}
__device__ __forceinline__ float funkey(unsigned int k) {
    return __uint_as_float((k & 0x80000000u) ? (k ^ 0x80000000u) : ~k);
}

__global__ void zero_f32_kernel(float* p, long long n) {
    long long i = (long long)blockIdx.x * blockDim.x + threadIdx.x;
    if (i < n) p[i] = 0.f;
}

__global__ void init_mkey_kernel(unsigned int* mkey, int n) {
    int i = blockIdx.x * blockDim.x + threadIdx.x;
    if (i < n) mkey[i] = fkey(-3.0e38f);
}

__global__ void init_bias_rows_kernel(float* out, const float* __restrict__ bo,
                                      long long total, int F) {
    long long i = (long long)blockIdx.x * blockDim.x + threadIdx.x;
    if (i < total) out[i] = bo[(int)(i % F)];
}

// ------------------------- GATv2 attention kernels -------------------------
// Edge list: e < E -> (src=ei[e], tgt=ei[E+e]); e >= E -> self loop (e-E).

__global__ void edge_score_kernel(const float* __restrict__ xl,
                                  const float* __restrict__ xr,
                                  const float* __restrict__ att,
                                  const int* __restrict__ ei,
                                  int E, int Nn, int H, int C,
                                  float* __restrict__ ep,
                                  unsigned int* __restrict__ mkey)
{
    int idx = blockIdx.x * blockDim.x + threadIdx.x;
    int ET = E + Nn;
    if (idx >= ET * H) return;
    int e = idx / H, h = idx - e * H;
    int src = (e < E) ? ei[e]     : (e - E);
    int tgt = (e < E) ? ei[E + e] : (e - E);
    int HC = H * C;
    const float* pl = xl + (size_t)src * HC + h * C;
    const float* pr = xr + (size_t)tgt * HC + h * C;
    const float* pa = att + h * C;
    float acc = 0.f;
    for (int c0 = 0; c0 < C; ++c0) {
        float v = pr[c0] + pl[c0];
        v = (v > 0.f) ? v : 0.2f * v;    // leaky_relu(0.2)
        acc += pa[c0] * v;
    }
    ep[idx] = acc;
    atomicMax(&mkey[(size_t)tgt * H + h], fkey(acc));
}

__global__ void edge_exp_kernel(float* __restrict__ ep,
                                const unsigned int* __restrict__ mkey,
                                float* __restrict__ ssum,
                                const int* __restrict__ ei,
                                int E, int Nn, int H)
{
    int idx = blockIdx.x * blockDim.x + threadIdx.x;
    int ET = E + Nn;
    if (idx >= ET * H) return;
    int e = idx / H, h = idx - e * H;
    int tgt = (e < E) ? ei[E + e] : (e - E);
    float m = funkey(mkey[(size_t)tgt * H + h]);
    float p = __expf(ep[idx] - m);
    ep[idx] = p;
    atomicAdd(&ssum[(size_t)tgt * H + h], p);
}

__global__ void edge_agg_kernel(const float* __restrict__ ep,
                                const float* __restrict__ ssum,
                                const float* __restrict__ xl,
                                const int* __restrict__ ei,
                                int E, int Nn, int H, int C,
                                float* __restrict__ out)
{
    int idx = blockIdx.x * blockDim.x + threadIdx.x;
    int ET = E + Nn;
    if (idx >= ET * H) return;
    int e = idx / H, h = idx - e * H;
    int src = (e < E) ? ei[e]     : (e - E);
    int tgt = (e < E) ? ei[E + e] : (e - E);
    int HC = H * C;
    float alpha = ep[idx] / ssum[(size_t)tgt * H + h];
    const float* pl = xl + (size_t)src * HC + h * C;
    float* po = out + (size_t)tgt * HC + h * C;
    for (int c0 = 0; c0 < C; ++c0)
        atomicAdd(&po[c0], alpha * pl[c0]);
}

// -------------------- graph LayerNorm (mode='graph') -----------------------

__global__ void reduce_meanvar_kernel(const float* __restrict__ x,
                                      long long total, float* scal)
{
    __shared__ float sh1[256], sh2[256];
    float s1 = 0.f, s2 = 0.f;
    for (long long i = (long long)blockIdx.x * blockDim.x + threadIdx.x;
         i < total; i += (long long)gridDim.x * blockDim.x) {
        float v = x[i];
        s1 += v; s2 += v * v;
    }
    int t = threadIdx.x;
    sh1[t] = s1; sh2[t] = s2;
    __syncthreads();
    for (int s = 128; s > 0; s >>= 1) {
        if (t < s) { sh1[t] += sh1[t + s]; sh2[t] += sh2[t + s]; }
        __syncthreads();
    }
    if (t == 0) {
        atomicAdd(&scal[0], sh1[0]);
        atomicAdd(&scal[1], sh2[0]);
    }
}

__global__ void gnorm_apply_relu_kernel(float* __restrict__ xio,
                                        const float* __restrict__ g,
                                        const float* __restrict__ b,
                                        const float* __restrict__ scal,
                                        long long total, int F)
{
    long long i = (long long)blockIdx.x * blockDim.x + threadIdx.x;
    if (i >= total) return;
    float cnt = (float)total;
    float mu  = scal[0] / cnt;
    float var = scal[1] / cnt - mu * mu;
    float rs  = rsqrtf(var + 1e-5f);
    int col = (int)(i % F);
    float v = (xio[i] - mu) * rs * g[col] + b[col];
    xio[i] = (v > 0.f) ? v : 0.f;   // relu fused
}

// ----------------------------- graph mean ----------------------------------

__global__ void graph_sum_kernel(const float* __restrict__ z, int Nn,
                                 float* __restrict__ gsum)
{
    int c = threadIdx.x;   // blockDim.x == 64
    float s = 0.f;
    for (int n = blockIdx.x; n < Nn; n += gridDim.x)
        s += z[(size_t)n * 64 + c];
    atomicAdd(&gsum[c], s);
}

__global__ void graph_finish_kernel(const float* __restrict__ gsum,
                                    float* __restrict__ out, int Nn)
{
    int c = threadIdx.x;
    if (c < 64) out[c] = gsum[c] / (float)Nn;
}

// ------------------------------- host side ---------------------------------

extern "C" void kernel_launch(void* const* d_in, const int* in_sizes, int n_in,
                              void* d_out, int out_size, void* d_ws, size_t ws_size,
                              hipStream_t stream)
{
    const float* x  = (const float*)d_in[0];
    const int*   ei = (const int*)d_in[1];
    const int N = in_sizes[0] / 64;       // IN = 64
    const int E = in_sizes[1] / 2;

    const float *Wl1=(const float*)d_in[3],  *bl1=(const float*)d_in[4];
    const float *Wr1=(const float*)d_in[5],  *br1=(const float*)d_in[6];
    const float *att1=(const float*)d_in[7], *bo1=(const float*)d_in[8];
    const float *Wl2=(const float*)d_in[9],  *bl2=(const float*)d_in[10];
    const float *Wr2=(const float*)d_in[11], *br2=(const float*)d_in[12];
    const float *att2=(const float*)d_in[13],*bo2=(const float*)d_in[14];
    const float *Wres2=(const float*)d_in[15];
    const float *Wl3=(const float*)d_in[16], *bl3=(const float*)d_in[17];
    const float *Wr3=(const float*)d_in[18], *br3=(const float*)d_in[19];
    const float *att3=(const float*)d_in[20],*bo3=(const float*)d_in[21];
    const float *Wres3=(const float*)d_in[22];
    const float *ln1_g=(const float*)d_in[23], *ln1_b=(const float*)d_in[24];
    const float *ln2_g=(const float*)d_in[25], *ln2_b=(const float*)d_in[26];
    const float *ln3_g=(const float*)d_in[27], *ln3_b=(const float*)d_in[28];
    const float *enc_W=(const float*)d_in[29], *enc_b=(const float*)d_in[30];
    const float *rp_W =(const float*)d_in[31], *rp_b =(const float*)d_in[32];
    const float *d1_W =(const float*)d_in[33], *d1_b =(const float*)d_in[34];
    const float *d2_W =(const float*)d_in[35], *d2_b =(const float*)d_in[36];
    const float *d3_W =(const float*)d_in[37], *d3_b =(const float*)d_in[38];

    // ---------------- workspace layout (floats) ----------------
    float* ws = (float*)d_ws;
    const size_t big = (size_t)N * 256;
    const long long ET = (long long)E + N;
    float* BA = ws;
    float* BB = ws + big;
    float* BC = ws + 2 * big;
    float* ep = ws + 3 * big;
    unsigned int* mkey = (unsigned int*)(ep + (size_t)ET * 4);
    float* ssum = (float*)(mkey + (size_t)N * 4);
    float* scal = ssum + (size_t)N * 4;
    float* gsum = scal + 8;
    _Float16* packBase = (_Float16*)(gsum + 64);   // packed f16 weights (~172K halves)

    float* z_out  = (float*)d_out;                       // [N,64]
    float* xh_out = (float*)d_out + (size_t)N * 64;      // [N,64]
    float* zg_out = (float*)d_out + (size_t)N * 128;     // [64]

    // --- pack all weight matrices into WMMA fragment layout (once per call) ---
    size_t packOff = 0;
    auto pack = [&](const float* W, int K, int Fo) -> _Float16* {
        _Float16* dst = packBase + packOff;
        int total = K * Fo;
        packOff += (size_t)total;
        pack_weight_kernel<<<(total + 255) / 256, 256, 0, stream>>>(W, dst, K, Fo);
        return dst;
    };
    _Float16* pWl1 = pack(Wl1, 64, 64),   *pWr1 = pack(Wr1, 64, 64);
    _Float16* pWl2 = pack(Wl2, 64, 128),  *pWr2 = pack(Wr2, 64, 128);
    _Float16* pWres2 = pack(Wres2, 64, 128);
    _Float16* pWl3 = pack(Wl3, 128, 256), *pWr3 = pack(Wr3, 128, 256);
    _Float16* pWres3 = pack(Wres3, 128, 256);
    _Float16* pEnc = pack(enc_W, 256, 64);
    _Float16* pRp  = pack(rp_W, 64, 64);
    _Float16* pD1  = pack(d1_W, 64, 128);
    _Float16* pD2  = pack(d2_W, 128, 64);
    _Float16* pD3  = pack(d3_W, 64, 64);

    auto gemm = [&](const float* A, const _Float16* Wp, const float* bias,
                    float* C, int Nr, int K, int Fo, int flags) {
        dim3 grid((Nr + 63) / 64, Fo / 64);
        gemm_wmma_kernel<<<grid, 128, 0, stream>>>(A, Wp, bias, C, Nr, K, Fo, flags);
    };

    // One GATv2 layer. Output (post-norm, post-relu) lands in `xrout`.
    auto gat_layer = [&](const float* hin, int Fin, int H, int C,
                         const _Float16* Wl, const float* bl,
                         const _Float16* Wr, const float* br,
                         const float* att, const float* bo, const _Float16* Wres,
                         float* xl, float* xrout,
                         const float* lng, const float* lnb) {
        const int HC = H * C;
        gemm(hin, Wl, bl, xl, N, Fin, HC, 0);      // source transform
        gemm(hin, Wr, br, xrout, N, Fin, HC, 0);   // target transform

        const int nh = N * H;
        init_mkey_kernel<<<(nh + 255) / 256, 256, 0, stream>>>(mkey, nh);
        zero_f32_kernel<<<(nh + 255) / 256, 256, 0, stream>>>(ssum, nh);

        const int eth = (int)(ET * H);
        edge_score_kernel<<<(eth + 255) / 256, 256, 0, stream>>>(
            xl, xrout, att, ei, E, N, H, C, ep, mkey);
        edge_exp_kernel<<<(eth + 255) / 256, 256, 0, stream>>>(
            ep, mkey, ssum, ei, E, N, H);

        // xrout is no longer needed as xr: reuse it as the output accumulator.
        const long long tot = (long long)N * HC;
        init_bias_rows_kernel<<<(int)((tot + 255) / 256), 256, 0, stream>>>(
            xrout, bo, tot, HC);
        edge_agg_kernel<<<(eth + 255) / 256, 256, 0, stream>>>(
            ep, ssum, xl, ei, E, N, H, C, xrout);

        if (Wres)
            gemm(hin, Wres, nullptr, xrout, N, Fin, HC, GEMM_ACCUM);

        // graph LayerNorm over all N*HC elements, then ReLU.
        zero_f32_kernel<<<1, 256, 0, stream>>>(scal, 2);
        reduce_meanvar_kernel<<<512, 256, 0, stream>>>(xrout, tot, scal);
        gnorm_apply_relu_kernel<<<(int)((tot + 255) / 256), 256, 0, stream>>>(
            xrout, lng, lnb, scal, tot, HC);
    };

    // conv1: H=1, C=64  (in 64)  -> h1 in BB
    gat_layer(x,  64, 1, 64, pWl1, bl1, pWr1, br1, att1, bo1, nullptr, BA, BB, ln1_g, ln1_b);
    // conv2: H=4, C=32  (in 64)  -> h2 in BC (residual)
    gat_layer(BB, 64, 4, 32, pWl2, bl2, pWr2, br2, att2, bo2, pWres2,  BA, BC, ln2_g, ln2_b);
    // conv3: H=4, C=64  (in 128) -> h3 in BB (residual)
    gat_layer(BC, 128, 4, 64, pWl3, bl3, pWr3, br3, att3, bo3, pWres3, BA, BB, ln3_g, ln3_b);

    // encoder: z = relu(h3 @ enc_W + enc_b) + (x @ rp_W + rp_b)
    gemm(BB, pEnc, enc_b, z_out, N, 256, 64, GEMM_RELU);
    gemm(x,  pRp,  rp_b,  z_out, N, 64,  64, GEMM_ACCUM);

    // z_graph = mean over nodes of z
    zero_f32_kernel<<<1, 64, 0, stream>>>(gsum, 64);
    graph_sum_kernel<<<256, 64, 0, stream>>>(z_out, N, gsum);
    graph_finish_kernel<<<1, 64, 0, stream>>>(gsum, zg_out, N);

    // decoder MLP
    gemm(z_out, pD1, d1_b, BA, N, 64, 128, GEMM_RELU);
    gemm(BA,    pD2, d2_b, BC, N, 128, 64, GEMM_RELU);
    gemm(BC,    pD3, d3_b, xh_out, N, 64, 64, 0);
}
